// SignedGCN_20512763805795
// MI455X (gfx1250) — matrix operand order repaired and checked by reading, not compiled, not run on hardware
//
#include <hip/hip_runtime.h>

// ---------------------------------------------------------------------------
// SignedGCN on MI455X (gfx1250, wave32).
// Edge aggregation = native fp32 global atomics (global_atomic_add_f32 via
// inline CDNA5 asm, no-return form -> STOREcnt, fire-and-forget) into
// L2-resident accumulators.
// Dense layers = v_wmma_f32_16x16x32_f16, one wave per 16-row node tile.
// ---------------------------------------------------------------------------

#define N_NODES 16896
#define IN_CH   64
#define HDIM    32

typedef __attribute__((ext_vector_type(16))) _Float16 v16h;
typedef __attribute__((ext_vector_type(8)))  float    v8f;

#define WMMA_F16(A, B, C) \
  __builtin_amdgcn_wmma_f32_16x16x32_f16(false, (A), false, (B), (short)0, (C), false, false)

// Native fp32 scatter-add: guarantee GLOBAL_ATOMIC_ADD_F32 (no CAS loop).
__device__ __forceinline__ void atomic_add_f32(float* p, float v) {
#if defined(__gfx1250__)
  asm volatile("global_atomic_add_f32 %0, %1, off" : : "v"(p), "v"(v) : "memory");
#else
  atomicAdd(p, v);
#endif
}

// ----- WMMA fragment helpers (layouts per CDNA5 ISA 7.12.2) -----------------

// A (16x32 f16): lane l owns row rowBase + (l&15).
// lanes 0-15:  elems 0..7 -> K=kBase+0..7,  elems 8..15 -> K=kBase+16..23
// lanes 16-31: elems 0..7 -> K=kBase+8..15, elems 8..15 -> K=kBase+24..31
__device__ __forceinline__ v16h load_A(const float* __restrict__ M, int rowBase, int ld,
                                       int kBase, float scale, int lane) {
  const int row = rowBase + (lane & 15);
  const int kHi = (lane >> 4) & 1;
  const float* p = M + row * ld + kBase + kHi * 8;
  v16h a;
#pragma unroll
  for (int j = 0; j < 8; ++j) {
    a[j]     = (_Float16)(p[j] * scale);
    a[8 + j] = (_Float16)(p[16 + j] * scale);
  }
  return a;
}

// B (32x16 f16) from row-major KxN weight: lane l owns column nBase + (l&15);
// lanes 0-15: elems 0..7 -> K=kBase+0..7, 8..15 -> K=kBase+16..23; lanes 16-31: +8.
__device__ __forceinline__ v16h load_B(const float* __restrict__ W, int kBase, int nBase,
                                       int ld, int lane) {
  const int col = nBase + (lane & 15);
  const int k0  = kBase + ((lane >> 4) & 1) * 8;
  v16h b;
#pragma unroll
  for (int j = 0; j < 16; ++j) b[j] = (_Float16)W[(k0 + (j < 8 ? j : j + 8)) * ld + col];
  return b;
}

// C/D (16x16 f32): VGPR r -> row rowBase + r + (lane>=16 ? 8 : 0), col = colBase + (l&15).
__device__ __forceinline__ void store_relu(float* __restrict__ out, int rowBase, int ld,
                                           int colBase, const float* __restrict__ bias,
                                           int biasBase, v8f c, int lane) {
  const int   col = colBase + (lane & 15);
  const float bv  = bias[biasBase + (lane & 15)];
  const int   r0  = ((lane >> 4) & 1) * 8;
#pragma unroll
  for (int r = 0; r < 8; ++r) {
    float v = c[r] + bv;
    out[(rowBase + r0 + r) * ld + col] = v > 0.0f ? v : 0.0f;
  }
}

// ----- Kernels --------------------------------------------------------------

__global__ void sgcn_zero_kernel(float* __restrict__ p, int n) {
  for (int i = blockIdx.x * blockDim.x + threadIdx.x; i < n; i += gridDim.x * blockDim.x)
    p[i] = 0.0f;
}

__global__ void sgcn_degree_kernel(const int* __restrict__ ei, int E, float* __restrict__ deg) {
  int e = blockIdx.x * blockDim.x + threadIdx.x;
  if (e < E) atomic_add_f32(&deg[ei[E + e]], 1.0f);  // dst row = ei[1][e]
}

// 16 threads/edge, 4 channels each: coalesced float4 gather, 4 fp32 atomics.
__global__ void sgcn_scatter64_kernel(const int* __restrict__ ei, int E,
                                      const float* __restrict__ feat,
                                      float* __restrict__ acc) {
  int t = blockIdx.x * blockDim.x + threadIdx.x;
  if (t >= E * 16) return;
  int e = t >> 4, q = t & 15;
  int src = ei[e], dst = ei[E + e];
  if (q == 0 && e + 1024 < E)  // warm L2/L0 for an upcoming source row
    __builtin_prefetch(feat + ei[e + 1024] * IN_CH, 0, 0);
  const float4 v = *(const float4*)(feat + src * IN_CH + q * 4);
  float* p = acc + dst * IN_CH + q * 4;
  atomic_add_f32(p + 0, v.x); atomic_add_f32(p + 1, v.y);
  atomic_add_f32(p + 2, v.z); atomic_add_f32(p + 3, v.w);
}

// 8 threads/edge: two 32-ch features from z scattered into two accumulators.
__global__ void sgcn_scatter32x2_kernel(const int* __restrict__ ei, int E,
                                        const float* __restrict__ z, int offA, int offB,
                                        float* __restrict__ accA, float* __restrict__ accB) {
  int t = blockIdx.x * blockDim.x + threadIdx.x;
  if (t >= E * 8) return;
  int e = t >> 3, q = t & 7;
  int src = ei[e], dst = ei[E + e];
  if (q == 0 && e + 1024 < E)
    __builtin_prefetch(z + ei[e + 1024] * 64, 0, 0);
  const float4 va = *(const float4*)(z + src * 64 + offA + q * 4);
  const float4 vb = *(const float4*)(z + src * 64 + offB + q * 4);
  float* pa = accA + dst * HDIM + q * 4;
  float* pb = accB + dst * HDIM + q * 4;
  atomic_add_f32(pa + 0, va.x); atomic_add_f32(pa + 1, va.y);
  atomic_add_f32(pa + 2, va.z); atomic_add_f32(pa + 3, va.w);
  atomic_add_f32(pb + 0, vb.x); atomic_add_f32(pb + 1, vb.y);
  atomic_add_f32(pb + 2, vb.z); atomic_add_f32(pb + 3, vb.w);
}

// Layer 1: z = relu([agg_p/deg_p @ W1_pl + h @ W1_pr + b1_pr,
//                    agg_n/deg_n @ W1_nl + h @ W1_nr + b1_nr])
__global__ void sgcn_layer1_kernel(const float* __restrict__ h,
                                   const float* __restrict__ agg_pos,
                                   const float* __restrict__ agg_neg,
                                   const float* __restrict__ deg_pos,
                                   const float* __restrict__ deg_neg,
                                   const float* __restrict__ W1_pl, const float* __restrict__ W1_pr,
                                   const float* __restrict__ b1_pr,
                                   const float* __restrict__ W1_nl, const float* __restrict__ W1_nr,
                                   const float* __restrict__ b1_nr,
                                   float* __restrict__ z) {
  const int lane = threadIdx.x & 31;
  const int wave = threadIdx.x >> 5;
  const int tile = blockIdx.x * (blockDim.x >> 5) + wave;
  const int rowBase = tile * 16;
  if (rowBase >= N_NODES) return;  // uniform per wave -> EXEC stays all-ones

  const int   row = rowBase + (lane & 15);
  const float rdp = 1.0f / fmaxf(deg_pos[row], 1.0f);
  const float rdn = 1.0f / fmaxf(deg_neg[row], 1.0f);

  v16h Ap0 = load_A(agg_pos, rowBase, IN_CH, 0,  rdp,  lane);
  v16h Ap1 = load_A(agg_pos, rowBase, IN_CH, 32, rdp,  lane);
  v16h An0 = load_A(agg_neg, rowBase, IN_CH, 0,  rdn,  lane);
  v16h An1 = load_A(agg_neg, rowBase, IN_CH, 32, rdn,  lane);
  v16h Ah0 = load_A(h,       rowBase, IN_CH, 0,  1.0f, lane);
  v16h Ah1 = load_A(h,       rowBase, IN_CH, 32, 1.0f, lane);

#pragma unroll
  for (int nt = 0; nt < 2; ++nt) {
    const int nb = nt * 16;
    v8f c = {};
    c = WMMA_F16(Ap0, load_B(W1_pl, 0,  nb, HDIM, lane), c);
    c = WMMA_F16(Ap1, load_B(W1_pl, 32, nb, HDIM, lane), c);
    c = WMMA_F16(Ah0, load_B(W1_pr, 0,  nb, HDIM, lane), c);
    c = WMMA_F16(Ah1, load_B(W1_pr, 32, nb, HDIM, lane), c);
    store_relu(z, rowBase, 64, nb, b1_pr, nb, c, lane);

    v8f d = {};
    d = WMMA_F16(An0, load_B(W1_nl, 0,  nb, HDIM, lane), d);
    d = WMMA_F16(An1, load_B(W1_nl, 32, nb, HDIM, lane), d);
    d = WMMA_F16(Ah0, load_B(W1_nr, 0,  nb, HDIM, lane), d);
    d = WMMA_F16(Ah1, load_B(W1_nr, 32, nb, HDIM, lane), d);
    store_relu(z, rowBase, 64, 32 + nb, b1_nr, nb, d, lane);
  }
}

// Layer 2: out = relu([ [a/dp, b/dn] @ W2_pl + zp @ W2_pr + b2_pr,
//                       [c/dp, d/dn] @ W2_nl + zn @ W2_nr + b2_nr])
__global__ void sgcn_layer2_kernel(const float* __restrict__ z,
                                   const float* __restrict__ agg_a, const float* __restrict__ agg_b,
                                   const float* __restrict__ agg_c, const float* __restrict__ agg_d,
                                   const float* __restrict__ deg_pos,
                                   const float* __restrict__ deg_neg,
                                   const float* __restrict__ W2_pl, const float* __restrict__ W2_pr,
                                   const float* __restrict__ b2_pr,
                                   const float* __restrict__ W2_nl, const float* __restrict__ W2_nr,
                                   const float* __restrict__ b2_nr,
                                   float* __restrict__ out) {
  const int lane = threadIdx.x & 31;
  const int wave = threadIdx.x >> 5;
  const int tile = blockIdx.x * (blockDim.x >> 5) + wave;
  const int rowBase = tile * 16;
  if (rowBase >= N_NODES) return;

  const int   row = rowBase + (lane & 15);
  const float rdp = 1.0f / fmaxf(deg_pos[row], 1.0f);
  const float rdn = 1.0f / fmaxf(deg_neg[row], 1.0f);

  v16h Apc0 = load_A(agg_a, rowBase, HDIM, 0,  rdp,  lane);  // pos_cat K=0..31
  v16h Apc1 = load_A(agg_b, rowBase, HDIM, 0,  rdn,  lane);  // pos_cat K=32..63
  v16h Anc0 = load_A(agg_c, rowBase, HDIM, 0,  rdp,  lane);  // neg_cat K=0..31
  v16h Anc1 = load_A(agg_d, rowBase, HDIM, 0,  rdn,  lane);  // neg_cat K=32..63
  v16h Azp  = load_A(z,     rowBase, 64,   0,  1.0f, lane);  // zp
  v16h Azn  = load_A(z,     rowBase, 64,   32, 1.0f, lane);  // zn

#pragma unroll
  for (int nt = 0; nt < 2; ++nt) {
    const int nb = nt * 16;
    v8f c = {};
    c = WMMA_F16(Apc0, load_B(W2_pl, 0,  nb, HDIM, lane), c);
    c = WMMA_F16(Apc1, load_B(W2_pl, 32, nb, HDIM, lane), c);
    c = WMMA_F16(Azp,  load_B(W2_pr, 0,  nb, HDIM, lane), c);
    store_relu(out, rowBase, 64, nb, b2_pr, nb, c, lane);

    v8f d = {};
    d = WMMA_F16(Anc0, load_B(W2_nl, 0,  nb, HDIM, lane), d);
    d = WMMA_F16(Anc1, load_B(W2_nl, 32, nb, HDIM, lane), d);
    d = WMMA_F16(Azn,  load_B(W2_nr, 0,  nb, HDIM, lane), d);
    store_relu(out, rowBase, 64, 32 + nb, b2_nr, nb, d, lane);
  }
}

// ----- Host launch ----------------------------------------------------------

extern "C" void kernel_launch(void* const* d_in, const int* in_sizes, int n_in,
                              void* d_out, int out_size, void* d_ws, size_t ws_size,
                              hipStream_t stream) {
  const int*   pos_ei  = (const int*)d_in[1];
  const int*   neg_ei  = (const int*)d_in[2];
  const float* x_param = (const float*)d_in[3];
  const float* W1_pl   = (const float*)d_in[4];
  const float* W1_pr   = (const float*)d_in[5];
  const float* b1_pr   = (const float*)d_in[6];
  const float* W1_nl   = (const float*)d_in[7];
  const float* W1_nr   = (const float*)d_in[8];
  const float* b1_nr   = (const float*)d_in[9];
  const float* W2_pl   = (const float*)d_in[10];
  const float* W2_pr   = (const float*)d_in[11];
  const float* b2_pr   = (const float*)d_in[12];
  const float* W2_nl   = (const float*)d_in[13];
  const float* W2_nr   = (const float*)d_in[14];
  const float* b2_nr   = (const float*)d_in[15];
  float* out = (float*)d_out;

  const int Ep = in_sizes[1] / 2;
  const int En = in_sizes[2] / 2;
  const int N  = N_NODES;

  // Workspace layout: [zero-init region | z]
  float* deg_pos  = (float*)d_ws;            // N
  float* deg_neg  = deg_pos  + N;            // N
  float* agg1_pos = deg_neg  + N;            // N*64
  float* agg1_neg = agg1_pos + N * IN_CH;    // N*64
  float* agg2_a   = agg1_neg + N * IN_CH;    // N*32
  float* agg2_b   = agg2_a   + N * HDIM;
  float* agg2_c   = agg2_b   + N * HDIM;
  float* agg2_d   = agg2_c   + N * HDIM;
  float* z        = agg2_d   + N * HDIM;     // N*64 (fully overwritten, no zeroing)
  const int nZero = (int)(z - deg_pos);      // 258*N floats

  const int BT = 256;

  sgcn_zero_kernel<<<2048, BT, 0, stream>>>(deg_pos, nZero);

  sgcn_degree_kernel<<<(Ep + BT - 1) / BT, BT, 0, stream>>>(pos_ei, Ep, deg_pos);
  sgcn_degree_kernel<<<(En + BT - 1) / BT, BT, 0, stream>>>(neg_ei, En, deg_neg);

  sgcn_scatter64_kernel<<<(Ep * 16 + BT - 1) / BT, BT, 0, stream>>>(pos_ei, Ep, x_param, agg1_pos);
  sgcn_scatter64_kernel<<<(En * 16 + BT - 1) / BT, BT, 0, stream>>>(neg_ei, En, x_param, agg1_neg);

  const int tileBlocks = (N / 16 + 7) / 8;   // 8 waves (tiles) per 256-thread block
  sgcn_layer1_kernel<<<tileBlocks, BT, 0, stream>>>(
      x_param, agg1_pos, agg1_neg, deg_pos, deg_neg,
      W1_pl, W1_pr, b1_pr, W1_nl, W1_nr, b1_nr, z);

  // pos edges: zp -> agg2_a, zn -> agg2_c ; neg edges: zn -> agg2_b, zp -> agg2_d
  sgcn_scatter32x2_kernel<<<(Ep * 8 + BT - 1) / BT, BT, 0, stream>>>(
      pos_ei, Ep, z, /*offA=*/0,  /*offB=*/32, agg2_a, agg2_c);
  sgcn_scatter32x2_kernel<<<(En * 8 + BT - 1) / BT, BT, 0, stream>>>(
      neg_ei, En, z, /*offA=*/32, /*offB=*/0,  agg2_b, agg2_d);

  sgcn_layer2_kernel<<<tileBlocks, BT, 0, stream>>>(
      z, agg2_a, agg2_b, agg2_c, agg2_d, deg_pos, deg_neg,
      W2_pl, W2_pr, b2_pr, W2_nl, W2_nr, b2_nr, out);
}
